// HMLSTMCell3_6657199309452
// MI455X (gfx1250) — compile-verified, matrix operand order checked by reading, not run on hardware
//
#include <hip/hip_runtime.h>
#include <hip/hip_bf16.h>

// ---------------------------------------------------------------------------
// HM-LSTM cell for MI455X (gfx1250): wave32, v_wmma_f32_16x16x32_bf16,
// TDM (tensor_load_to_lds) double-buffered LDS staging, BK=64 K-steps.
//
// s = (h_bottom*z_bottom)@W + h@R + (h_top*z)@U + b   -> fused K=3072 GEMM,
// 4 gate column tiles per workgroup, LSTM epilogue in-register.
// ---------------------------------------------------------------------------

typedef __attribute__((ext_vector_type(16))) __bf16 v16bf;
typedef __attribute__((ext_vector_type(8)))  float  v8f;
typedef unsigned int u32x4 __attribute__((ext_vector_type(4)));
typedef int          i32x4 __attribute__((ext_vector_type(4)));
typedef int          i32x8 __attribute__((ext_vector_type(8)));

#define B_DIM  4096
#define H_DIM  1024
#define K_DIM  3072            // 3 * 1024
#define NW     4097            // 4*H + 1
#define BK     64              // K per staged step (2 WMMA k-phases)
#define LDSS   80              // LDS row stride (bf16 elems): 128B row + 32B pad
#define TM     128             // rows per WG
#define TN     64              // cols per gate per WG (256 total)
#define NSTEP  (K_DIM / BK)    // 48

#if __has_builtin(__builtin_amdgcn_tensor_load_to_lds) && \
    __has_builtin(__builtin_amdgcn_s_wait_tensorcnt)
#define USE_TDM 1
#else
#define USE_TDM 0
#endif

__device__ __forceinline__ float sigf(float x) {
    return 1.0f / (1.0f + __expf(-x));
}
__device__ __forceinline__ float tanhfast(float x) {
    float e = __expf(2.0f * x);
    return 1.0f - 2.0f / (e + 1.0f);       // saturates cleanly at +/-1
}

#if USE_TDM
// ---------------------------------------------------------------------------
// Tensor Data Mover: load a (t2 x t1 x t0) tile of a row-major tensor into
// LDS with 32B padding per 128B row (pad_interval=32 DW, pad_amount=8 DW).
// Dims/strides in elements (data_size = 2 bytes). ISA D# layout, §8.3-8.6.
// ---------------------------------------------------------------------------
__device__ __forceinline__ void tdm_load(
    unsigned ldsOff, unsigned long long ga /*bytes*/,
    unsigned td0, unsigned td1,
    unsigned long long st0, unsigned long long st1,
    unsigned td2, unsigned t0, unsigned t1, unsigned t2)
{
    u32x4 g0;
    g0[0] = 1u;                                            // count=1 (user D#)
    g0[1] = ldsOff;                                        // lds_addr
    g0[2] = (unsigned)ga;                                  // global_addr lo
    g0[3] = (unsigned)((ga >> 32) & 0x1FFFFFFull) | (2u << 30); // addr hi | type=2

    i32x8 g1;
    g1[0] = (int)((1u << 16)      // data_size = 2B
                | (1u << 20)      // pad_enable
                | (4u << 22)      // pad_interval: 32 DWORDs (128B)
                | (7u << 25));    // pad_amount:   8 DWORDs (32B)
    g1[1] = (int)((td0 & 0xFFFFu) << 16);                  // tensor_dim0[15:0]
    g1[2] = (int)((td0 >> 16) | ((td1 & 0xFFFFu) << 16));  // dim0 hi | dim1 lo
    g1[3] = (int)((td1 >> 16) | (t0 << 16));               // dim1 hi | tile_dim0
    g1[4] = (int)(t1 | (t2 << 16));                        // tile_dim1 | tile_dim2
    g1[5] = (int)(unsigned)st0;                            // dim0_stride[31:0]
    g1[6] = (int)((unsigned)((st0 >> 32) & 0xFFFFull)      // dim0_stride[47:32]
                | (((unsigned)(st1 & 0xFFFFull)) << 16));  // dim1_stride[15:0]
    g1[7] = (int)(unsigned)((st1 >> 16) & 0xFFFFFFFFull);  // dim1_stride[47:16]

    i32x4 g2;
    g2[0] = (int)td2;  g2[1] = 0;  g2[2] = 0;  g2[3] = 0;  // tensor_dim2
    i32x4 g3 = {0, 0, 0, 0};

#if defined(__clang_major__) && __clang_major__ >= 23
    i32x8 g4 = {0, 0, 0, 0, 0, 0, 0, 0};
    __builtin_amdgcn_tensor_load_to_lds(g0, g1, g2, g3, g4, 0);
#else
    __builtin_amdgcn_tensor_load_to_lds(g0, g1, g2, g3, 0);
#endif
}
#endif  // USE_TDM

// ---------------------------------------------------------------------------
// Pack A = [h_bottom*z_bottom | h | h_top*z] into bf16, row-major 4096x3072.
// ---------------------------------------------------------------------------
__global__ void pack_A(const float* __restrict__ hb, const float* __restrict__ h,
                       const float* __restrict__ ht, const float* __restrict__ z,
                       const float* __restrict__ zb, __bf16* __restrict__ A)
{
    const int k = blockIdx.x * 256 + threadIdx.x;   // 0..3071
    const int b = blockIdx.y;                       // 0..4095
    float v;
    if (k < 1024)       v = hb[(size_t)b * 1024 + k]          * zb[b];
    else if (k < 2048)  v = h [(size_t)b * 1024 + (k - 1024)];
    else                v = ht[(size_t)b * 1024 + (k - 2048)] * z[b];
    A[(size_t)b * K_DIM + k] = (__bf16)v;
}

// ---------------------------------------------------------------------------
// Pack Wt[n][k] = {W|R|U}[k][n] as bf16, N-major 4096x3072 (tiled transpose).
// ---------------------------------------------------------------------------
__global__ void pack_W(const float* __restrict__ W, const float* __restrict__ R,
                       const float* __restrict__ U, __bf16* __restrict__ Wt)
{
    __shared__ float tile[32][33];
    const int tx = threadIdx.x, ty = threadIdx.y;   // 32 x 8
    const int k0 = blockIdx.x * 32;
    const int n0 = blockIdx.y * 32;
#pragma unroll
    for (int i = 0; i < 4; ++i) {
        const int kk = k0 + ty + i * 8;
        const float* src; int kr;
        if (kk < 1024)      { src = W; kr = kk;        }
        else if (kk < 2048) { src = R; kr = kk - 1024; }
        else                { src = U; kr = kk - 2048; }
        tile[ty + i * 8][tx] = src[(size_t)kr * NW + n0 + tx];
    }
    __syncthreads();
#pragma unroll
    for (int i = 0; i < 4; ++i) {
        const int nn = n0 + ty + i * 8;
        Wt[(size_t)nn * K_DIM + k0 + tx] = (__bf16)tile[tx][ty + i * 8];
    }
}

// ---------------------------------------------------------------------------
// Main fused GEMM + LSTM epilogue.
// Grid: (H/64, B/128). Block: 256 = 8 wave32s (4 row-waves x 2 col-waves).
// WG tile: 128 rows x 64 cols x 4 gates. Wave: 32x32x4 -> 16 v8f accums.
// Double-buffered LDS (BK=64); tiles staged by the Tensor Data Mover (wave 0).
// ---------------------------------------------------------------------------
__global__ __launch_bounds__(256)
void hmlstm_gemm(const __bf16* __restrict__ A, const __bf16* __restrict__ Wt,
                 const float* __restrict__ bias, const float* __restrict__ c_in,
                 const float* __restrict__ h_in, const float* __restrict__ z,
                 const float* __restrict__ zb, float* __restrict__ out)
{
    __shared__ alignas(32) __bf16 sA[2][TM * LDSS];        // 2 x 20 KB
    __shared__ alignas(32) __bf16 sB[2][4 * TN * LDSS];    // 2 x 40 KB

    const int tid  = threadIdx.x;
    const int lane = tid & 31;
    const int wv   = tid >> 5;
    const int wr   = wv & 3;               // 4 row-waves, 32 rows each
    const int wc   = wv >> 2;              // 2 col-waves, 32 cols each
    const int l16  = lane & 15;
    const int khalf = (lane >> 4) << 4;    // 0 or 16 elems
    const int row0 = blockIdx.y * TM;
    const int n0   = blockIdx.x * TN;

    v8f acc[4][2][2] = {};                 // [gate][nt][mt]

#if USE_TDM
    const unsigned ldsA[2] = { (unsigned)(size_t)&sA[0][0], (unsigned)(size_t)&sA[1][0] };
    const unsigned ldsB[2] = { (unsigned)(size_t)&sB[0][0], (unsigned)(size_t)&sB[1][0] };
    const unsigned long long gA0 = (unsigned long long)(size_t)(A + (size_t)row0 * K_DIM);
    const unsigned long long gB0 = (unsigned long long)(size_t)(Wt + (size_t)n0 * K_DIM);
    if (wv == 0) {
        // A: 2D tile 128x64 of 4096x3072 tensor
        tdm_load(ldsA[0], gA0, K_DIM, B_DIM, K_DIM, 0ull, 0, BK, TM, 0);
        // B: 3D tile [4 gates][64 n][64 k]; gate plane stride = 1024*3072
        tdm_load(ldsB[0], gB0, K_DIM, H_DIM, K_DIM,
                 (unsigned long long)H_DIM * K_DIM, 4, BK, TN, 4);
    }
#else
    uint4 rgA[4], rgB[8];
#pragma unroll
    for (int i = 0; i < 4; ++i) {
        const int idx = tid + i * 256, r = idx >> 3, q = (idx & 7) << 3;
        rgA[i] = *(const uint4*)(A + (size_t)(row0 + r) * K_DIM + q);
    }
#pragma unroll
    for (int i = 0; i < 8; ++i) {
        const int idx = tid + i * 256, r = idx >> 3, q = (idx & 7) << 3;
        const int wtRow = ((r >> 6) << 10) + n0 + (r & 63);
        rgB[i] = *(const uint4*)(Wt + (size_t)wtRow * K_DIM + q);
    }
#pragma unroll
    for (int i = 0; i < 4; ++i) {
        const int idx = tid + i * 256, r = idx >> 3, q = (idx & 7) << 3;
        *(uint4*)(&sA[0][r * LDSS + q]) = rgA[i];
    }
#pragma unroll
    for (int i = 0; i < 8; ++i) {
        const int idx = tid + i * 256, r = idx >> 3, q = (idx & 7) << 3;
        *(uint4*)(&sB[0][r * LDSS + q]) = rgB[i];
    }
#endif

    for (int it = 0; it < NSTEP; ++it) {
        const int cur = it & 1;
        const int nxt = cur ^ 1;
        const int knext = (it + 1) * BK;

#if USE_TDM
        __builtin_amdgcn_s_wait_tensorcnt(0);   // scalar; no-op for non-issuing waves
#endif
        __syncthreads();                        // buf[cur] published

#if USE_TDM
        if (wv == 0 && it + 1 < NSTEP) {
            tdm_load(ldsA[nxt], gA0 + (size_t)knext * sizeof(__bf16),
                     K_DIM, B_DIM, K_DIM, 0ull, 0, BK, TM, 0);
            tdm_load(ldsB[nxt], gB0 + (size_t)knext * sizeof(__bf16),
                     K_DIM, H_DIM, K_DIM,
                     (unsigned long long)H_DIM * K_DIM, 4, BK, TN, 4);
        }
#else
        if (it + 1 < NSTEP) {
#pragma unroll
            for (int i = 0; i < 4; ++i) {
                const int idx = tid + i * 256, r = idx >> 3, q = (idx & 7) << 3;
                rgA[i] = *(const uint4*)(A + (size_t)(row0 + r) * K_DIM + knext + q);
            }
#pragma unroll
            for (int i = 0; i < 8; ++i) {
                const int idx = tid + i * 256, r = idx >> 3, q = (idx & 7) << 3;
                const int wtRow = ((r >> 6) << 10) + n0 + (r & 63);
                rgB[i] = *(const uint4*)(Wt + (size_t)wtRow * K_DIM + knext + q);
            }
        }
#endif

        // ---- 2 k-phases x 16 WMMA per wave from buf[cur] ----
        const __bf16* pA = &sA[cur][0];
        const __bf16* pB = &sB[cur][0];
#pragma unroll
        for (int kb = 0; kb < 2; ++kb) {
            const int ko = kb * 32 + khalf;
            v16bf afrag[2];
            v16bf bfrag[4][2];
#pragma unroll
            for (int mt = 0; mt < 2; ++mt)
                afrag[mt] = *(const v16bf*)(pA + (wr * 32 + mt * 16 + l16) * LDSS + ko);
#pragma unroll
            for (int g = 0; g < 4; ++g)
#pragma unroll
                for (int nt = 0; nt < 2; ++nt)
                    bfrag[g][nt] = *(const v16bf*)
                        (pB + (g * TN + wc * 32 + nt * 16 + l16) * LDSS + ko);
#pragma unroll
            for (int g = 0; g < 4; ++g)
#pragma unroll
                for (int nt = 0; nt < 2; ++nt)
#pragma unroll
                    for (int mt = 0; mt < 2; ++mt)
                        acc[g][nt][mt] = __builtin_amdgcn_wmma_f32_16x16x32_bf16(
                            false, afrag[mt], false, bfrag[g][nt],
                            (short)0, acc[g][nt][mt], false, false);
        }

#if !USE_TDM
        if (it + 1 < NSTEP) {
#pragma unroll
            for (int i = 0; i < 4; ++i) {
                const int idx = tid + i * 256, r = idx >> 3, q = (idx & 7) << 3;
                *(uint4*)(&sA[nxt][r * LDSS + q]) = rgA[i];
            }
#pragma unroll
            for (int i = 0; i < 8; ++i) {
                const int idx = tid + i * 256, r = idx >> 3, q = (idx & 7) << 3;
                *(uint4*)(&sB[nxt][r * LDSS + q]) = rgB[i];
            }
        }
#endif
    }

    // ---- epilogue: C/D layout lane = N%16, VGPR v = M (+8 for lanes 16-31) ----
    const int mbase = (lane >> 4) << 3;
#pragma unroll
    for (int nt = 0; nt < 2; ++nt) {
        const int j = n0 + wc * 32 + nt * 16 + l16;         // 0..1023
        const float bi  = bias[j];
        const float bg  = bias[H_DIM + j];
        const float bo  = bias[2 * H_DIM + j];
        const float bf_ = bias[3 * H_DIM + j];
#pragma unroll
        for (int mt = 0; mt < 2; ++mt) {
#pragma unroll
            for (int v = 0; v < 8; ++v) {
                const int row = row0 + wr * 32 + mt * 16 + mbase + v;
                const float zr  = z[row];
                const float zbr = zb[row];
                const size_t off = (size_t)row * H_DIM + j;
                const float cr = c_in[off];
                const float hr = h_in[off];

                const float i_ = sigf(acc[0][nt][mt][v] + bi);
                const float g_ = tanhfast(acc[1][nt][mt][v] + bg);
                const float o_ = sigf(acc[2][nt][mt][v] + bo);
                const float f_ = sigf(acc[3][nt][mt][v] + bf_);

                const float ig  = i_ * g_;
                const float omz = 1.0f - zr;
                const float cn  = zr * ig + omz * (1.0f - zbr) * cr
                                + omz * zbr * (f_ * cr + ig);
                const float tc  = tanhfast(cn);
                const float hn  = (zr + omz * zbr) * o_ * tc
                                + omz * (1.0f - zbr) * hr;

                out[off] = hn;                                // h_new
                out[(size_t)B_DIM * H_DIM + off] = cn;        // c_new
            }
        }
    }
}

// ---------------------------------------------------------------------------
// z-column: sz[b] = dot with column 4096 of W/R/U + bias[4096]; hard threshold.
// ---------------------------------------------------------------------------
__global__ void hmlstm_zcol(const float* __restrict__ hb, const float* __restrict__ h,
                            const float* __restrict__ ht, const float* __restrict__ z,
                            const float* __restrict__ zb, const float* __restrict__ W,
                            const float* __restrict__ R, const float* __restrict__ U,
                            const float* __restrict__ bias, float* __restrict__ out_z)
{
    __shared__ float red[256];
    const int b = blockIdx.x;
    const int tid = threadIdx.x;
    const float zr = z[b], zbr = zb[b];
    float s = 0.0f;
    for (int k = tid; k < 1024; k += 256) {
        const size_t a = (size_t)b * 1024 + k;
        const size_t w = (size_t)k * NW + (NW - 1);
        s += hb[a] * zbr * W[w];
        s += h [a] *       R[w];
        s += ht[a] * zr  * U[w];
    }
    red[tid] = s;
    __syncthreads();
    for (int st = 128; st > 0; st >>= 1) {
        if (tid < st) red[tid] += red[tid + st];
        __syncthreads();
    }
    if (tid == 0) {
        const float sz = red[0] + bias[NW - 1];
        float zt = (7.0f * sz + 1.0f) * 0.5f;
        zt = fminf(fmaxf(zt, 0.0f), 1.0f);
        out_z[b] = (zt > 0.5f) ? 1.0f : 0.0f;
    }
}

// ---------------------------------------------------------------------------
extern "C" void kernel_launch(void* const* d_in, const int* in_sizes, int n_in,
                              void* d_out, int out_size, void* d_ws, size_t ws_size,
                              hipStream_t stream)
{
    const float* hb   = (const float*)d_in[0];
    const float* h    = (const float*)d_in[1];
    const float* ht   = (const float*)d_in[2];
    const float* c    = (const float*)d_in[3];
    const float* z    = (const float*)d_in[4];
    const float* zb   = (const float*)d_in[5];
    const float* W    = (const float*)d_in[6];
    const float* R    = (const float*)d_in[7];
    const float* U    = (const float*)d_in[8];
    const float* bias = (const float*)d_in[9];

    float* out = (float*)d_out;   // [h_new | c_new | z_new]

    __bf16* Apack = (__bf16*)d_ws;
    __bf16* Wt    = Apack + (size_t)B_DIM * K_DIM;

    pack_A<<<dim3(K_DIM / 256, B_DIM), 256, 0, stream>>>(hb, h, ht, z, zb, Apack);
    pack_W<<<dim3(K_DIM / 32, 4 * H_DIM / 32), dim3(32, 8), 0, stream>>>(W, R, U, Wt);
    hmlstm_gemm<<<dim3(H_DIM / TN, B_DIM / TM), 256, 0, stream>>>(
        Apack, Wt, bias, c, h, z, zb, out);
    hmlstm_zcol<<<B_DIM, 256, 0, stream>>>(
        hb, h, ht, z, zb, W, R, U, bias, out + (size_t)2 * B_DIM * H_DIM);
}
